// Net_4217657885096
// MI455X (gfx1250) — compile-verified, matrix operand order checked by reading, not run on hardware
//
#include <hip/hip_runtime.h>
#include <hip/hip_bf16.h>
#include <math.h>

// ---------------------------------------------------------------------------
// Types for WMMA (CDNA5 / gfx1250, wave32)
// ---------------------------------------------------------------------------
typedef __attribute__((ext_vector_type(16))) _Float16 v16h;
typedef __attribute__((ext_vector_type(8)))  _Float16 v8h;
typedef __attribute__((ext_vector_type(8)))  float    v8f;

#define BGR 128      // number of graphs
#define EPSV 1e-5f

// ---------------------------------------------------------------------------
// Utility fills
// ---------------------------------------------------------------------------
__global__ void fill_f32(float* __restrict__ p, float v, long long n) {
    long long i = (long long)blockIdx.x * blockDim.x + threadIdx.x;
    if (i < n) p[i] = v;
}

__global__ void fill_i32(int* __restrict__ p, int v, long long n) {
    long long i = (long long)blockIdx.x * blockDim.x + threadIdx.x;
    if (i < n) p[i] = v;
}

// ---------------------------------------------------------------------------
// f32 [N,K] -> f16 [N,Kp] zero-padded (row-major), coalesced.
// ---------------------------------------------------------------------------
__global__ void cvt_x_f16(const float* __restrict__ x, _Float16* __restrict__ xh,
                          long long NKp, int K, int Kp) {
    long long i = (long long)blockIdx.x * blockDim.x + threadIdx.x;
    if (i >= NKp) return;
    long long r = i / Kp;
    int k = (int)(i % Kp);
    xh[i] = (k < K) ? (_Float16)x[r * K + k] : (_Float16)0.f;
}

// f32 W [K,Cout] -> f16 Wt [Cout,Kp] (transposed + zero-padded) so B-fragment
// gathers are contiguous 128-bit loads.
__global__ void cvt_w_f16t(const float* __restrict__ W, _Float16* __restrict__ Wt,
                           int K, int Kp, int Cout) {
    int i = blockIdx.x * blockDim.x + threadIdx.x;
    if (i >= Cout * Kp) return;
    int c = i / Kp;
    int k = i % Kp;
    Wt[i] = (k < K) ? (_Float16)W[(size_t)k * Cout + c] : (_Float16)0.f;
}

// ---------------------------------------------------------------------------
// Edge scatter: agg[dst] += x[src] (masked). One thread per (edge, channel).
// Atomics resolve in L2 (192MB covers the whole working set).
// ---------------------------------------------------------------------------
__global__ void scatter_add(const float* __restrict__ x,
                            const int* __restrict__ src,
                            const int* __restrict__ dst,
                            const int* __restrict__ mask,
                            long long EC, int C,
                            float* __restrict__ agg) {
    long long t = (long long)blockIdx.x * blockDim.x + threadIdx.x;
    if (t >= EC) return;
    int e = (int)(t / C);
    int c = (int)(t % C);
    if (mask && !mask[e]) return;
    atomicAdd(&agg[(size_t)dst[e] * C + c], x[(size_t)src[e] * C + c]);
}

// ---------------------------------------------------------------------------
// Fused dual GEMM: Y = X@Wr + AGG@Wn + bias  via v_wmma_f32_16x16x32_f16.
// Inputs are pre-converted/padded f16: Xh/Ah [N,KP] row-major, Wrt/Wnt
// [COUT,KP] (W transposed). KP/COUT are compile-time so every loop fully
// unrolls: fragment arrays are statically indexed registers (no v_movrels),
// no bounds checks, EXEC stays all-1s around the WMMAs.
//
// wave32 16-bit fragment layout (ISA 7.12.2): per lane the 16 f16 elements
// cover K = kb..kb+7 then kb+16..kb+23 (kb = 8*(lane>=16)) -> exactly two
// contiguous, 16B-aligned 8xf16 loads per fragment.
// ---------------------------------------------------------------------------
#define SHUF16(lo, hi) __builtin_shufflevector(lo, hi, 0,1,2,3,4,5,6,7,8,9,10,11,12,13,14,15)

template<int KP, int COUT>
__global__ void gconv_gemm(const _Float16* __restrict__ Xh,
                           const _Float16* __restrict__ Ah,
                           const _Float16* __restrict__ Wrt,
                           const _Float16* __restrict__ Wnt,
                           const float* __restrict__ bias,
                           float* __restrict__ Y,
                           int N) {
    constexpr int KTN    = KP >> 5;    // 1 or 2 k-steps of 32
    constexpr int NTILES = COUT >> 4;  // 2 or 4 column tiles

    const int lane = threadIdx.x & 31;
    const int wv   = threadIdx.x >> 5;
    const int mt   = blockIdx.x * 4 + wv;
    const int mtiles = N >> 4;
    if (mt >= mtiles) return;                 // uniform per wave: EXEC stays all-1s

    const int hlf = lane >> 4;                // 0 or 1
    const int l15 = lane & 15;
    const int m0  = mt << 4;
    const int kb  = hlf << 3;                 // K base offset per half-wave
    const int arow = m0 + l15;

    // A fragments are invariant across output-column tiles: load once.
    v16h a1[KTN], a2[KTN];
#pragma unroll
    for (int kt = 0; kt < KTN; ++kt) {
        const size_t rb = (size_t)arow * KP + (kt << 5) + kb;
        v8h lo = *(const v8h*)(Xh + rb);
        v8h hi = *(const v8h*)(Xh + rb + 16);
        a1[kt] = SHUF16(lo, hi);
        lo = *(const v8h*)(Ah + rb);
        hi = *(const v8h*)(Ah + rb + 16);
        a2[kt] = SHUF16(lo, hi);
    }

#pragma unroll
    for (int nt = 0; nt < NTILES; ++nt) {
        const int col = (nt << 4) + l15;
        v8f acc = {};
#pragma unroll
        for (int kt = 0; kt < KTN; ++kt) {
            const size_t wb = (size_t)col * KP + (kt << 5) + kb;
            v8h lo = *(const v8h*)(Wrt + wb);
            v8h hi = *(const v8h*)(Wrt + wb + 16);
            v16h b1 = SHUF16(lo, hi);
            lo = *(const v8h*)(Wnt + wb);
            hi = *(const v8h*)(Wnt + wb + 16);
            v16h b2 = SHUF16(lo, hi);
            acc = __builtin_amdgcn_wmma_f32_16x16x32_f16(
                false, a1[kt], false, b1, (short)0, acc, false, false);
            acc = __builtin_amdgcn_wmma_f32_16x16x32_f16(
                false, a2[kt], false, b2, (short)0, acc, false, false);
        }
        const float bv = bias[col];
#pragma unroll
        for (int r = 0; r < 8; ++r) {
            int orow = m0 + (hlf << 3) + r;
            Y[(size_t)orow * COUT + col] = acc[r] + bv;
        }
    }
}

// ---------------------------------------------------------------------------
// BatchNorm (training mode, biased variance) over node dimension.
// ---------------------------------------------------------------------------
__global__ void bn_stats(const float* __restrict__ h, int N, int C,
                         float* __restrict__ stats /* [0:C)=mean [C:2C)=istd */) {
    const int c = blockIdx.x;
    const int t = threadIdx.x;
    __shared__ float s1[256], s2[256];
    float sum = 0.f, sq = 0.f;
    for (int i = t; i < N; i += 256) {
        float v = h[(size_t)i * C + c];
        sum += v; sq += v * v;
    }
    s1[t] = sum; s2[t] = sq;
    __syncthreads();
    for (int o = 128; o > 0; o >>= 1) {
        if (t < o) { s1[t] += s1[t + o]; s2[t] += s2[t + o]; }
        __syncthreads();
    }
    if (t == 0) {
        float m = s1[0] / (float)N;
        float v = s2[0] / (float)N - m * m;
        stats[c]     = m;
        stats[C + c] = rsqrtf(v + EPSV);
    }
}

__global__ void bn_relu(float* __restrict__ h, long long NC, int C,
                        const float* __restrict__ stats,
                        const float* __restrict__ gam,
                        const float* __restrict__ bet) {
    long long i = (long long)blockIdx.x * blockDim.x + threadIdx.x;
    if (i >= NC) return;
    int c = (int)(i % C);
    float v = h[i];
    v = (v - stats[c]) * stats[C + c] * gam[c] + bet[c];
    h[i] = fmaxf(v, 0.f);
}

// ---------------------------------------------------------------------------
// TopK pooling: one block per graph, blockDim == n (power of two <= 1024).
// Bitonic sort in LDS, descending score, lower-index wins ties (lax.top_k).
// ---------------------------------------------------------------------------
__global__ void topk_pool(const float* __restrict__ h,
                          const float* __restrict__ p,
                          float* __restrict__ xn,
                          int* __restrict__ remap,
                          int n, int k) {
    const int g = blockIdx.x;
    const int t = threadIdx.x;
    __shared__ float ss[1024];
    __shared__ int   si[1024];

    float nrm = 0.f;
#pragma unroll
    for (int j = 0; j < 64; ++j) { float pv = p[j]; nrm += pv * pv; }
    nrm = sqrtf(nrm);

    const float* row = h + ((size_t)g * n + t) * 64;
    float sc = 0.f;
#pragma unroll
    for (int j = 0; j < 64; ++j) sc += row[j] * p[j];
    ss[t] = sc / nrm;
    si[t] = t;
    __syncthreads();

    for (int ksz = 2; ksz <= n; ksz <<= 1) {
        for (int j = ksz >> 1; j > 0; j >>= 1) {
            int ixj = t ^ j;
            if (ixj > t) {
                bool dirDesc = ((t & ksz) == 0);
                float a = ss[t], b = ss[ixj];
                int ia = si[t], ib = si[ixj];
                // true if element at ixj ranks BEFORE element at t (descending)
                bool bFirst = (b > a) || (b == a && ib < ia);
                if (bFirst == dirDesc) {
                    ss[t] = b; si[t] = ib;
                    ss[ixj] = a; si[ixj] = ia;
                }
            }
            __syncthreads();
        }
    }

    if (t < k) {
        int old = si[t];
        float th = tanhf(ss[t]);
        const float* srow = h  + ((size_t)g * n + old) * 64;
        float*       drow = xn + ((size_t)g * k + t)   * 64;
#pragma unroll
        for (int c = 0; c < 64; ++c) drow[c] = srow[c] * th;
        remap[g * n + old] = g * k + t;
    }
}

__global__ void edge_remap(const int* __restrict__ src,
                           const int* __restrict__ dst,
                           const int* __restrict__ mask,
                           const int* __restrict__ remap,
                           int E,
                           int* __restrict__ nsrc,
                           int* __restrict__ ndst,
                           int* __restrict__ nmask) {
    int e = blockIdx.x * blockDim.x + threadIdx.x;
    if (e >= E) return;
    int m  = mask ? mask[e] : 1;
    int ns = remap[src[e]];
    int nd = remap[dst[e]];
    nmask[e] = (m && ns >= 0 && nd >= 0) ? 1 : 0;
    nsrc[e]  = ns < 0 ? 0 : ns;
    ndst[e]  = nd < 0 ? 0 : nd;
}

// ---------------------------------------------------------------------------
// Readout: concat(max-pool, mean-pool) accumulated into rsum (r1+r2+r3+r4).
// One block per graph, 64 threads (one per channel).
// ---------------------------------------------------------------------------
__global__ void readout_add(const float* __restrict__ h,
                            float* __restrict__ rsum, int k) {
    int g = blockIdx.x;
    int c = threadIdx.x;   // 64
    float mx = -INFINITY, sm = 0.f;
    for (int i = 0; i < k; ++i) {
        float v = h[((size_t)g * k + i) * 64 + c];
        mx = fmaxf(mx, v);
        sm += v;
    }
    rsum[g * 128 + c]      += mx;
    rsum[g * 128 + 64 + c] += sm / (float)k;
}

// ---------------------------------------------------------------------------
// MLP head: 128 -> 64 (relu) -> 32 (relu) -> 10 -> log_softmax.
// One block per graph, 64 threads.
// ---------------------------------------------------------------------------
__global__ void mlp_head(const float* __restrict__ r,
                         const float* __restrict__ Wl1, const float* __restrict__ bl1,
                         const float* __restrict__ Wl2, const float* __restrict__ bl2,
                         const float* __restrict__ Wl3, const float* __restrict__ bl3,
                         float* __restrict__ out) {
    int g = blockIdx.x, t = threadIdx.x;
    __shared__ float h1[64], h2[32], lg[10], red[2];
    {
        float a = bl1[t];
        for (int j = 0; j < 128; ++j) a += r[g * 128 + j] * Wl1[j * 64 + t];
        h1[t] = fmaxf(a, 0.f);
    }
    __syncthreads();
    if (t < 32) {
        float a = bl2[t];
        for (int j = 0; j < 64; ++j) a += h1[j] * Wl2[j * 32 + t];
        h2[t] = fmaxf(a, 0.f);
    }
    __syncthreads();
    if (t < 10) {
        float a = bl3[t];
        for (int j = 0; j < 32; ++j) a += h2[j] * Wl3[j * 10 + t];
        lg[t] = a;
    }
    __syncthreads();
    if (t == 0) {
        float m = -INFINITY;
        for (int i = 0; i < 10; ++i) m = fmaxf(m, lg[i]);
        float s = 0.f;
        for (int i = 0; i < 10; ++i) s += expf(lg[i] - m);
        red[0] = m; red[1] = logf(s);
    }
    __syncthreads();
    if (t < 10) out[g * 10 + t] = lg[t] - red[0] - red[1];
}

// ---------------------------------------------------------------------------
// Host orchestration
// ---------------------------------------------------------------------------
extern "C" void kernel_launch(void* const* d_in, const int* in_sizes, int n_in,
                              void* d_out, int out_size, void* d_ws, size_t ws_size,
                              hipStream_t stream) {
    (void)n_in; (void)out_size; (void)ws_size;

    const float* x    = (const float*)d_in[0];
    const int*   esrc = (const int*)d_in[1];
    const int*   edst = (const int*)d_in[2];
    const int    E    = in_sizes[1];

    const float* Wr1  = (const float*)d_in[3];
    const float* Wn1  = (const float*)d_in[4];
    const float* b1   = (const float*)d_in[5];
    const float* g1   = (const float*)d_in[6];
    const float* bt1  = (const float*)d_in[7];
    const float* Wr1b = (const float*)d_in[8];
    const float* Wn1b = (const float*)d_in[9];
    const float* b1b  = (const float*)d_in[10];
    const float* g1b  = (const float*)d_in[11];
    const float* bt1b = (const float*)d_in[12];
    const float* p1   = (const float*)d_in[13];
    const float* Wr2  = (const float*)d_in[14];
    const float* Wn2  = (const float*)d_in[15];
    const float* b2   = (const float*)d_in[16];
    const float* g2   = (const float*)d_in[17];
    const float* bt2  = (const float*)d_in[18];
    const float* p2   = (const float*)d_in[19];
    const float* Wr3  = (const float*)d_in[20];
    const float* Wn3  = (const float*)d_in[21];
    const float* b3   = (const float*)d_in[22];
    const float* g3   = (const float*)d_in[23];
    const float* bt3  = (const float*)d_in[24];
    const float* p3   = (const float*)d_in[25];
    const float* Wr4  = (const float*)d_in[26];
    const float* Wn4  = (const float*)d_in[27];
    const float* b4   = (const float*)d_in[28];
    const float* g4   = (const float*)d_in[29];
    const float* bt4  = (const float*)d_in[30];
    const float* p4   = (const float*)d_in[31];
    const float* Wl1  = (const float*)d_in[32];
    const float* bl1  = (const float*)d_in[33];
    const float* Wl2  = (const float*)d_in[34];
    const float* bl2  = (const float*)d_in[35];
    const float* Wl3  = (const float*)d_in[36];
    const float* bl3  = (const float*)d_in[37];

    // --- workspace carve-out (256B aligned) ---
    char* wp = (char*)d_ws;
    auto alloc = [&](size_t bytes) -> void* {
        void* p = (void*)wp;
        wp += (bytes + 255) & ~(size_t)255;
        return p;
    };
    const int N0 = BGR * 1024;                 // 131072
    float* agg  = (float*)alloc((size_t)N0 * 64 * 4);   // f32 scatter target, reused
    _Float16* xh = (_Float16*)alloc((size_t)N0 * 64 * 2); // padded f16 activations
    _Float16* ah = (_Float16*)alloc((size_t)N0 * 64 * 2); // padded f16 aggregates
    _Float16* wrh = (_Float16*)alloc((size_t)64 * 64 * 2); // Wr^T f16
    _Float16* wnh = (_Float16*)alloc((size_t)64 * 64 * 2); // Wn^T f16
    float* hA   = (float*)alloc((size_t)N0 * 32 * 4);
    float* hB   = (float*)alloc((size_t)N0 * 64 * 4);
    float* hP1  = (float*)alloc((size_t)BGR * 512 * 64 * 4);
    float* hC   = (float*)alloc((size_t)BGR * 512 * 64 * 4);
    float* hP2  = (float*)alloc((size_t)BGR * 256 * 64 * 4);
    float* hD   = (float*)alloc((size_t)BGR * 256 * 64 * 4);
    float* hP3  = (float*)alloc((size_t)BGR * 128 * 64 * 4);
    float* hE   = (float*)alloc((size_t)BGR * 128 * 64 * 4);
    float* hP4  = (float*)alloc((size_t)BGR * 39  * 64 * 4);
    float* rsum = (float*)alloc((size_t)BGR * 128 * 4);
    float* stats= (float*)alloc(128 * 4);
    int* remap  = (int*)alloc((size_t)N0 * 4);
    int* s1 = (int*)alloc((size_t)E * 4); int* d1 = (int*)alloc((size_t)E * 4); int* m1 = (int*)alloc((size_t)E * 4);
    int* s2 = (int*)alloc((size_t)E * 4); int* d2 = (int*)alloc((size_t)E * 4); int* m2 = (int*)alloc((size_t)E * 4);
    int* s3 = (int*)alloc((size_t)E * 4); int* d3 = (int*)alloc((size_t)E * 4); int* m3 = (int*)alloc((size_t)E * 4);
    int* s4 = (int*)alloc((size_t)E * 4); int* d4 = (int*)alloc((size_t)E * 4); int* m4 = (int*)alloc((size_t)E * 4);

    auto blocks = [](long long n, int bs) { return (unsigned)((n + bs - 1) / bs); };

    auto gconv_bn = [&](const float* hin, int N, int K, int Cout,
                        const int* src, const int* dst, const int* msk,
                        const float* Wr, const float* Wn, const float* bb,
                        const float* gg, const float* bt, float* hout) {
        const int Kp = (K <= 32) ? 32 : 64;
        // 1) aggregate neighbor features (f32 atomics)
        long long NK = (long long)N * K;
        fill_f32<<<blocks(NK, 256), 256, 0, stream>>>(agg, 0.f, NK);
        long long EC = (long long)E * K;
        scatter_add<<<blocks(EC, 256), 256, 0, stream>>>(hin, src, dst, msk, EC, K, agg);
        // 2) convert/pad operands to f16
        long long NKp = (long long)N * Kp;
        cvt_x_f16<<<blocks(NKp, 256), 256, 0, stream>>>(hin, xh, NKp, K, Kp);
        cvt_x_f16<<<blocks(NKp, 256), 256, 0, stream>>>(agg, ah, NKp, K, Kp);
        cvt_w_f16t<<<blocks(Cout * Kp, 256), 256, 0, stream>>>(Wr, wrh, K, Kp, Cout);
        cvt_w_f16t<<<blocks(Cout * Kp, 256), 256, 0, stream>>>(Wn, wnh, K, Kp, Cout);
        // 3) dual WMMA GEMM + bias (compile-time shapes -> fully unrolled)
        unsigned gb = blocks(N / 16, 4);
        if (Kp == 32 && Cout == 32)
            gconv_gemm<32, 32><<<gb, 128, 0, stream>>>(xh, ah, wrh, wnh, bb, hout, N);
        else if (Kp == 32 && Cout == 64)
            gconv_gemm<32, 64><<<gb, 128, 0, stream>>>(xh, ah, wrh, wnh, bb, hout, N);
        else
            gconv_gemm<64, 64><<<gb, 128, 0, stream>>>(xh, ah, wrh, wnh, bb, hout, N);
        // 4) BatchNorm + ReLU
        bn_stats<<<Cout, 256, 0, stream>>>(hout, N, Cout, stats);
        long long NC = (long long)N * Cout;
        bn_relu<<<blocks(NC, 256), 256, 0, stream>>>(hout, NC, Cout, stats, gg, bt);
    };

    auto topk = [&](const float* hin, int n, int k, const float* p,
                    const int* src, const int* dst, const int* msk,
                    float* xout, int* nsrc, int* ndst, int* nmsk) {
        long long Bn = (long long)BGR * n;
        fill_i32<<<blocks(Bn, 256), 256, 0, stream>>>(remap, -1, Bn);
        topk_pool<<<BGR, n, 0, stream>>>(hin, p, xout, remap, n, k);
        edge_remap<<<blocks(E, 256), 256, 0, stream>>>(src, dst, msk, remap, E, nsrc, ndst, nmsk);
    };

    // r accumulator
    fill_f32<<<blocks((long long)BGR * 128, 256), 256, 0, stream>>>(rsum, 0.f, (long long)BGR * 128);

    // Layer 1: 3 -> 32, then 32 -> 64 (same full edge set, all edges valid)
    gconv_bn(x,  N0, 3,  32, esrc, edst, nullptr, Wr1,  Wn1,  b1,  g1,  bt1,  hA);
    gconv_bn(hA, N0, 32, 64, esrc, edst, nullptr, Wr1b, Wn1b, b1b, g1b, bt1b, hB);

    // Pool 1: 1024 -> 512, readout r1
    topk(hB, 1024, 512, p1, esrc, edst, nullptr, hP1, s1, d1, m1);
    readout_add<<<BGR, 64, 0, stream>>>(hP1, rsum, 512);

    // Layer 2 + pool 2: 512 -> 256
    gconv_bn(hP1, BGR * 512, 64, 64, s1, d1, m1, Wr2, Wn2, b2, g2, bt2, hC);
    topk(hC, 512, 256, p2, s1, d1, m1, hP2, s2, d2, m2);
    readout_add<<<BGR, 64, 0, stream>>>(hP2, rsum, 256);

    // Layer 3 + pool 3: 256 -> 128
    gconv_bn(hP2, BGR * 256, 64, 64, s2, d2, m2, Wr3, Wn3, b3, g3, bt3, hD);
    topk(hD, 256, 128, p3, s2, d2, m2, hP3, s3, d3, m3);
    readout_add<<<BGR, 64, 0, stream>>>(hP3, rsum, 128);

    // Layer 4 + pool 4: 128 -> 39
    gconv_bn(hP3, BGR * 128, 64, 64, s3, d3, m3, Wr4, Wn4, b4, g4, bt4, hE);
    topk(hE, 128, 39, p4, s3, d3, m3, hP4, s4, d4, m4);
    readout_add<<<BGR, 64, 0, stream>>>(hP4, rsum, 39);

    // MLP head + log_softmax
    mlp_head<<<BGR, 64, 0, stream>>>(rsum, Wl1, bl1, Wl2, bl2, Wl3, bl3, (float*)d_out);
}
